// Attention_86517821210894
// MI455X (gfx1250) — compile-verified, matrix operand order checked by reading, not run on hardware
//
#include <hip/hip_runtime.h>
#include <stdint.h>

#define B_  2
#define N_  4096
#define C_  512
#define H_  8
#define D_  64
#define NQKV 1536         // 3*C
#define EPS 1e-5f

typedef __attribute__((ext_vector_type(16))) __bf16   v16bf;
typedef __attribute__((ext_vector_type(8)))  float    v8f;
typedef __attribute__((ext_vector_type(8)))  uint32_t v8u;

__device__ __forceinline__ uint16_t f2bf(float f) {
    uint32_t u = __builtin_bit_cast(uint32_t, f);
    u += 0x7FFFu + ((u >> 16) & 1u);          // round-to-nearest-even
    return (uint16_t)(u >> 16);
}
__device__ __forceinline__ uint32_t pack2(float a, float b) {
    return (uint32_t)f2bf(a) | ((uint32_t)f2bf(b) << 16);
}
// two 128-bit loads -> one 8xu32 WMMA operand (compiler emits *_load_b128)
__device__ __forceinline__ v8u ld2x128(const uint16_t* p0, const uint16_t* p1) {
    uint4 a = *(const uint4*)p0;
    uint4 b = *(const uint4*)p1;
    v8u r;
    r[0] = a.x; r[1] = a.y; r[2] = a.z; r[3] = a.w;
    r[4] = b.x; r[5] = b.y; r[6] = b.z; r[7] = b.w;
    return r;
}
__device__ __forceinline__ float wave_red_sum(float v) {
    #pragma unroll
    for (int m = 16; m >= 1; m >>= 1) v += __shfl_xor(v, m, 32);
    return v;
}
// reductions within a 16-lane half (rows of a WMMA C tile live per-half)
__device__ __forceinline__ float half_red_max(float v) {
    #pragma unroll
    for (int m = 8; m >= 1; m >>= 1) v = fmaxf(v, __shfl_xor(v, m, 32));
    return v;
}
__device__ __forceinline__ float half_red_sum(float v) {
    #pragma unroll
    for (int m = 8; m >= 1; m >>= 1) v += __shfl_xor(v, m, 32);
    return v;
}

// ---------------------------------------------------------------------------
// Kernel 1: pre-LayerNorm + QKV projection (bf16 WMMA, fp32 accumulate)
// Block: 256 threads (8 waves), handles 16 tokens x all 1536 outputs.
// W slab staged in LDS as [n_local][k_pair] packed bf16 pairs -> B operand is
// two ds_load_b128 per WMMA. V stored transposed [B,H,D,N] so the attention
// PV B-operand becomes two global_load_b128 per WMMA.
// ---------------------------------------------------------------------------
__global__ __launch_bounds__(256)
void ln_qkv_kernel(const float* __restrict__ x,
                   const float* __restrict__ w,
                   const float* __restrict__ bias,
                   const float* __restrict__ g_pre,
                   const float* __restrict__ beta_pre,
                   float*       __restrict__ xn,
                   uint16_t*    __restrict__ qo,
                   uint16_t*    __restrict__ ko,
                   uint16_t*    __restrict__ vt) {
    __shared__ uint16_t a_lds[16][C_];        // 16 KB : normalized x, bf16
    __shared__ uint32_t wpair[384 * 16];      // 24 KB : W slab, [n][kpair]

    const int tid   = threadIdx.x;
    const int lane  = tid & 31;
    const int wv    = tid >> 5;               // wave id 0..7
    const int t0    = blockIdx.x * 16;        // first flat token (B*N space)
    const int m_row = lane & 15;
    const int hi    = lane >> 4;
    const int kb    = hi * 8;

    // ---- pre-LayerNorm: each wave normalizes 2 tokens ----
    #pragma unroll
    for (int s = 0; s < 2; ++s) {
        const int tl  = wv * 2 + s;           // 0..15
        const int tok = t0 + tl;
        const float* xr = x + (size_t)tok * C_;
        float vals[16];
        float sum = 0.f, sq = 0.f;
        #pragma unroll
        for (int i = 0; i < 4; ++i) {
            float4 f = ((const float4*)xr)[lane * 4 + i];
            vals[4*i+0] = f.x; vals[4*i+1] = f.y;
            vals[4*i+2] = f.z; vals[4*i+3] = f.w;
            sum += f.x + f.y + f.z + f.w;
            sq  += f.x*f.x + f.y*f.y + f.z*f.z + f.w*f.w;
        }
        sum = wave_red_sum(sum);
        sq  = wave_red_sum(sq);
        const float mean = sum * (1.f / C_);
        const float var  = sq  * (1.f / C_) - mean * mean;
        const float rstd = rsqrtf(var + EPS);
        float* xnr = xn + (size_t)tok * C_;
        #pragma unroll
        for (int i = 0; i < 16; ++i)
            vals[i] = (vals[i] - mean) * rstd * g_pre[lane * 16 + i] +
                      beta_pre[lane * 16 + i];
        // packed bf16 -> LDS, two ds_store_b128 per lane
        {
            uint32_t pk[8];
            #pragma unroll
            for (int i = 0; i < 8; ++i) pk[i] = pack2(vals[2*i], vals[2*i+1]);
            uint4* dst = (uint4*)&a_lds[tl][lane * 16];
            dst[0] = make_uint4(pk[0], pk[1], pk[2], pk[3]);
            dst[1] = make_uint4(pk[4], pk[5], pk[6], pk[7]);
        }
        #pragma unroll
        for (int i = 0; i < 4; ++i)
            ((float4*)xnr)[lane * 4 + i] =
                make_float4(vals[4*i], vals[4*i+1], vals[4*i+2], vals[4*i+3]);
    }
    __syncthreads();

    // fixed pair-row per thread for the cooperative slab loader (no division)
    const int pr = tid >> 4;            // pair-row 0..15
    const int c0 = tid & 15;            // base float4 column

    // ---- QKV GEMM: 4 passes over the 1536-wide output, 384 cols each ----
    for (int npass = 0; npass < 4; ++npass) {
        const int nbase = npass * 384;
        v8f acc[3];
        #pragma unroll
        for (int t = 0; t < 3; ++t)
            #pragma unroll
            for (int j = 0; j < 8; ++j) acc[t][j] = 0.f;

        for (int k0 = 0; k0 < C_; k0 += 32) {
            __syncthreads();                  // previous slab fully consumed
            // cooperative slab load: pack k-pairs along K into [n][kpair]
            const float* r0 = w + (size_t)(k0 + 2 * pr) * NQKV + nbase;
            #pragma unroll
            for (int j = 0; j < 6; ++j) {
                const int cc = c0 + j * 16;         // float4 col 0..95
                float4 f0 = ((const float4*)r0)[cc];
                float4 f1 = ((const float4*)(r0 + NQKV))[cc];
                const int nl = cc * 4;
                wpair[(nl + 0) * 16 + pr] = pack2(f0.x, f1.x);
                wpair[(nl + 1) * 16 + pr] = pack2(f0.y, f1.y);
                wpair[(nl + 2) * 16 + pr] = pack2(f0.z, f1.z);
                wpair[(nl + 3) * 16 + pr] = pack2(f0.w, f1.w);
            }
            if (k0 + 32 < C_)
                __builtin_prefetch(w + (size_t)(k0 + 32) * NQKV + nbase + tid, 0, 0);
            __syncthreads();

            // A operand: two ds_load_b128 (pairs contiguous along K)
            const v8u aw = ld2x128(&a_lds[m_row][k0 + kb],
                                   &a_lds[m_row][k0 + kb + 16]);
            const v16bf A = __builtin_bit_cast(v16bf, aw);

            #pragma unroll
            for (int t = 0; t < 3; ++t) {
                const int nl = (wv * 3 + t) * 16 + m_row;   // local col in slab
                const uint32_t* wb = &wpair[nl * 16 + (kb >> 1)];
                uint4 blo = *(const uint4*)wb;
                uint4 bhi = *(const uint4*)(wb + 8);
                v8u bw;
                bw[0] = blo.x; bw[1] = blo.y; bw[2] = blo.z; bw[3] = blo.w;
                bw[4] = bhi.x; bw[5] = bhi.y; bw[6] = bhi.z; bw[7] = bhi.w;
                const v16bf Bv = __builtin_bit_cast(v16bf, bw);
                acc[t] = __builtin_amdgcn_wmma_f32_16x16x32_bf16(
                    false, A, false, Bv, (short)0, acc[t], false, false);
            }
        }

        // ---- epilogue: bias add, scatter into q/k [B,H,N,D] and vT [B,H,D,N]
        #pragma unroll
        for (int t = 0; t < 3; ++t) {
            const int gn     = nbase + (wv * 3 + t) * 16 + m_row;
            const float bia  = bias[gn];
            const int which  = gn >> 9;
            const int within = gn & 511;
            const int hh     = within >> 6;
            const int dd     = within & 63;
            const int tok0   = t0 + hi * 8;       // 8 consecutive tokens
            const int bb     = tok0 >> 12;
            const int nn0    = tok0 & (N_ - 1);
            if (which == 2) {
                // V transposed: rows of the C tile are contiguous tokens ->
                // one 16-byte packed store per lane
                uint32_t p0 = pack2(acc[t][0] + bia, acc[t][1] + bia);
                uint32_t p1 = pack2(acc[t][2] + bia, acc[t][3] + bia);
                uint32_t p2 = pack2(acc[t][4] + bia, acc[t][5] + bia);
                uint32_t p3 = pack2(acc[t][6] + bia, acc[t][7] + bia);
                uint4* dst = (uint4*)&vt[(((size_t)bb * H_ + hh) * D_ + dd) * N_ + nn0];
                *dst = make_uint4(p0, p1, p2, p3);
            } else {
                uint16_t* outp = (which == 0) ? qo : ko;
                #pragma unroll
                for (int r = 0; r < 8; ++r)
                    outp[(((size_t)bb * H_ + hh) * N_ + (nn0 + r)) * D_ + dd] =
                        f2bf(acc[t][r] + bia);
            }
        }
    }
}

// ---------------------------------------------------------------------------
// Kernel 2: flash attention. Block: 128 threads (4 waves); each wave owns
// 16 query rows of one (b,h) head and streams keys in tiles of 64 (softmax
// bookkeeping amortized over 2x keys). All WMMA operands are b128 load pairs.
// ---------------------------------------------------------------------------
__global__ __launch_bounds__(128)
void flash_attn_kernel(const uint16_t* __restrict__ q,
                       const uint16_t* __restrict__ k,
                       const uint16_t* __restrict__ vt,   // [B,H,D,N]
                       float*          __restrict__ val) {
    __shared__ uint16_t plds[4][16][64];      // per-wave P transpose, 8 KB

    const int tid   = threadIdx.x;
    const int lane  = tid & 31;
    const int wv    = tid >> 5;
    const int bh    = blockIdx.y;             // b*H + h
    const int b     = bh >> 3;
    const int h     = bh & (H_ - 1);
    const int q0    = blockIdx.x * 64 + wv * 16;
    const int m_row = lane & 15;
    const int hi    = lane >> 4;
    const int kb    = hi * 8;

    const uint16_t* qh  = q  + (size_t)bh * N_ * D_;
    const uint16_t* kh  = k  + (size_t)bh * N_ * D_;
    const uint16_t* vth = vt + (size_t)bh * D_ * N_;

    // Q as A-operand: two 16x32 halves of the 16x64 query tile
    const uint16_t* qrow = qh + (size_t)(q0 + m_row) * D_;
    const v16bf Qlo = __builtin_bit_cast(v16bf, ld2x128(&qrow[kb],      &qrow[kb + 16]));
    const v16bf Qhi = __builtin_bit_cast(v16bf, ld2x128(&qrow[32 + kb], &qrow[48 + kb]));

    float m_run[8], l_run[8];
    v8f o[4];
    #pragma unroll
    for (int r = 0; r < 8; ++r) { m_run[r] = -1e30f; l_run[r] = 0.f; }
    #pragma unroll
    for (int t = 0; t < 4; ++t)
        #pragma unroll
        for (int r = 0; r < 8; ++r) o[t][r] = 0.f;

    for (int key0 = 0; key0 < N_; key0 += 64) {
        // ---- S = Q K^T for 64 keys (four 16-col tiles) ----
        v8f s[4];
        #pragma unroll
        for (int ct = 0; ct < 4; ++ct) {
            const uint16_t* krow = kh + (size_t)(key0 + ct * 16 + m_row) * D_;
            const v16bf Klo = __builtin_bit_cast(v16bf, ld2x128(&krow[kb],      &krow[kb + 16]));
            const v16bf Khi = __builtin_bit_cast(v16bf, ld2x128(&krow[32 + kb], &krow[48 + kb]));
            v8f c0;
            #pragma unroll
            for (int r = 0; r < 8; ++r) c0[r] = 0.f;
            c0 = __builtin_amdgcn_wmma_f32_16x16x32_bf16(
                false, Qlo, false, Klo, (short)0, c0, false, false);
            c0 = __builtin_amdgcn_wmma_f32_16x16x32_bf16(
                false, Qhi, false, Khi, (short)0, c0, false, false);
            s[ct] = c0;
        }

        // ---- online softmax update (fp32) ----
        #pragma unroll
        for (int r = 0; r < 8; ++r) {
            float e[4];
            #pragma unroll
            for (int ct = 0; ct < 4; ++ct) e[ct] = s[ct][r] * 0.125f;  // 1/sqrt(64)
            const float lmax = fmaxf(fmaxf(e[0], e[1]), fmaxf(e[2], e[3]));
            const float tmax = half_red_max(lmax);
            const float mn   = fmaxf(m_run[r], tmax);
            const float sc   = __expf(m_run[r] - mn);
            #pragma unroll
            for (int ct = 0; ct < 4; ++ct) {
                e[ct] = __expf(e[ct] - mn);
                s[ct][r] = e[ct];
            }
            const float rs = half_red_sum((e[0] + e[1]) + (e[2] + e[3]));
            l_run[r] = l_run[r] * sc + rs;
            m_run[r] = mn;
            #pragma unroll
            for (int t = 0; t < 4; ++t) o[t][r] *= sc;
        }

        // ---- transpose P (C-layout -> A-layout) through per-wave LDS ----
        #pragma unroll
        for (int r = 0; r < 8; ++r) {
            #pragma unroll
            for (int ct = 0; ct < 4; ++ct)
                plds[wv][r + 8 * hi][m_row + 16 * ct] = f2bf(s[ct][r]);
        }
        asm volatile("s_wait_dscnt 0" ::: "memory");
        const v16bf P0 = __builtin_bit_cast(v16bf,
            ld2x128(&plds[wv][m_row][kb],      &plds[wv][m_row][kb + 16]));
        const v16bf P1 = __builtin_bit_cast(v16bf,
            ld2x128(&plds[wv][m_row][32 + kb], &plds[wv][m_row][48 + kb]));

        // ---- O += P * V : V transposed -> pairs along keys contiguous,
        //      each B operand is two global_load_b128 ----
        #pragma unroll
        for (int t = 0; t < 4; ++t) {
            const int d = t * 16 + m_row;
            const uint16_t* vrow = vth + (size_t)d * N_ + key0;
            const v16bf V0 = __builtin_bit_cast(v16bf,
                ld2x128(&vrow[kb], &vrow[kb + 16]));
            o[t] = __builtin_amdgcn_wmma_f32_16x16x32_bf16(
                false, P0, false, V0, (short)0, o[t], false, false);
            const v16bf V1 = __builtin_bit_cast(v16bf,
                ld2x128(&vrow[32 + kb], &vrow[48 + kb]));
            o[t] = __builtin_amdgcn_wmma_f32_16x16x32_bf16(
                false, P1, false, V1, (short)0, o[t], false, false);
        }
    }

    // ---- epilogue: normalize by row sum, write fp32 [B,N,C] ----
    #pragma unroll
    for (int r = 0; r < 8; ++r) {
        const float inv = 1.f / l_run[r];
        const int tok = q0 + r + 8 * hi;
        float* orow = val + ((size_t)b * N_ + tok) * C_ + h * D_;
        #pragma unroll
        for (int t = 0; t < 4; ++t)
            orow[t * 16 + m_row] = o[t][r] * inv;
    }
}

// ---------------------------------------------------------------------------
// Kernel 3: post-LayerNorm + residual, in place over d_out.
// One wave per token.
// ---------------------------------------------------------------------------
__global__ __launch_bounds__(256)
void postln_residual_kernel(float* __restrict__ out,          // holds attn val
                            const float* __restrict__ xn,
                            const float* __restrict__ g_post,
                            const float* __restrict__ beta_post) {
    const int tid  = threadIdx.x;
    const int lane = tid & 31;
    const int wv   = tid >> 5;
    const int tok  = blockIdx.x * 8 + wv;

    float* vr = out + (size_t)tok * C_;
    const float* xr = xn + (size_t)tok * C_;

    float vals[16];
    float sum = 0.f, sq = 0.f;
    #pragma unroll
    for (int i = 0; i < 4; ++i) {
        float4 f = ((const float4*)vr)[lane * 4 + i];
        vals[4*i+0] = f.x; vals[4*i+1] = f.y;
        vals[4*i+2] = f.z; vals[4*i+3] = f.w;
        sum += f.x + f.y + f.z + f.w;
        sq  += f.x*f.x + f.y*f.y + f.z*f.z + f.w*f.w;
    }
    sum = wave_red_sum(sum);
    sq  = wave_red_sum(sq);
    const float mean = sum * (1.f / C_);
    const float var  = sq  * (1.f / C_) - mean * mean;
    const float rstd = rsqrtf(var + EPS);

    #pragma unroll
    for (int i = 0; i < 4; ++i) {
        const int c = lane * 16 + i * 4;
        float4 xf = ((const float4*)xr)[lane * 4 + i];
        float4 r;
        r.x = xf.x + (vals[4*i+0] - mean) * rstd * g_post[c+0] + beta_post[c+0];
        r.y = xf.y + (vals[4*i+1] - mean) * rstd * g_post[c+1] + beta_post[c+1];
        r.z = xf.z + (vals[4*i+2] - mean) * rstd * g_post[c+2] + beta_post[c+2];
        r.w = xf.w + (vals[4*i+3] - mean) * rstd * g_post[c+3] + beta_post[c+3];
        ((float4*)vr)[lane * 4 + i] = r;
    }
}

// ---------------------------------------------------------------------------
extern "C" void kernel_launch(void* const* d_in, const int* in_sizes, int n_in,
                              void* d_out, int out_size, void* d_ws, size_t ws_size,
                              hipStream_t stream) {
    (void)in_sizes; (void)n_in; (void)out_size; (void)ws_size;

    const float* x         = (const float*)d_in[0];
    const float* w_qkv     = (const float*)d_in[1];
    const float* b_qkv     = (const float*)d_in[2];
    const float* g_pre     = (const float*)d_in[3];
    const float* beta_pre  = (const float*)d_in[4];
    const float* g_post    = (const float*)d_in[5];
    const float* beta_post = (const float*)d_in[6];
    float* out = (float*)d_out;

    // workspace layout
    char* ws = (char*)d_ws;
    const size_t XN_BYTES  = (size_t)B_ * N_ * C_ * sizeof(float);     // 16 MB
    const size_t QKV_BYTES = (size_t)B_ * N_ * C_ * sizeof(uint16_t);  // 8 MB each
    float*    xn = (float*)ws;
    uint16_t* q  = (uint16_t*)(ws + XN_BYTES);
    uint16_t* k  = (uint16_t*)(ws + XN_BYTES + QKV_BYTES);
    uint16_t* vt = (uint16_t*)(ws + XN_BYTES + 2 * QKV_BYTES);

    // 1) pre-LN + QKV projection (V stored transposed)
    ln_qkv_kernel<<<dim3((B_ * N_) / 16), 256, 0, stream>>>(
        x, w_qkv, b_qkv, g_pre, beta_pre, xn, q, k, vt);

    // 2) flash attention -> writes fp32 attention output into d_out
    flash_attn_kernel<<<dim3(N_ / 64, B_ * H_), 128, 0, stream>>>(q, k, vt, out);

    // 3) post-LN + residual, in place on d_out
    postln_residual_kernel<<<dim3((B_ * N_) / 8), 256, 0, stream>>>(
        out, xn, g_post, beta_post);
}
